// D_Align_7834020348491
// MI455X (gfx1250) — matrix operand order repaired and checked by reading, not compiled, not run on hardware
//
#include <hip/hip_runtime.h>

// ---------- types ----------
typedef __attribute__((ext_vector_type(16))) __bf16 v16bf;
typedef __attribute__((ext_vector_type(8)))  float  v8f;
typedef __attribute__((ext_vector_type(4)))  unsigned int u32x4;
typedef unsigned int   u32;
typedef unsigned short u16;
typedef unsigned long long u64;

union Frag { v16bf v; u32x4 q[2]; };

__device__ __forceinline__ u16 f32_to_bf16(float f) {
  u32 u = __builtin_bit_cast(u32, f);
  u += 0x7fffu + ((u >> 16) & 1u);          // round-to-nearest-even
  return (u16)(u >> 16);
}
__device__ __forceinline__ float bf16lo(u32 u) { return __builtin_bit_cast(float, u << 16); }
__device__ __forceinline__ float bf16hi(u32 u) { return __builtin_bit_cast(float, u & 0xffff0000u); }

// ---------- gfx1250 async global->LDS copy (ASYNCcnt-tracked, bypasses VGPRs) ----------
#define USE_ASYNC_LDS 1

__device__ __forceinline__ void async_copy_b128(const void* gptr, void* lptr) {
#if USE_ASYNC_LDS
  // VDST = per-lane LDS byte offset (low 32 bits of generic LDS address),
  // VADDR = 64-bit global address, GV mode.
  u32 lds_off = (u32)(size_t)lptr;
  u64 gaddr   = (u64)(size_t)gptr;
  asm volatile("global_load_async_to_lds_b128 %0, %1, off"
               :: "v"(lds_off), "v"(gaddr) : "memory");
#else
  *(u32x4*)lptr = *(const u32x4*)gptr;
#endif
}
__device__ __forceinline__ void async_fence() {
#if USE_ASYNC_LDS
  asm volatile("s_wait_asynccnt 0x0" ::: "memory");
#endif
}

// ---------- constants (from reference) ----------
#define BATCH 4
#define NF    64
#define IMH   128
#define IMW   224
#define OFFCH 72   // 2*G_OFF*9

// ---------- weight pack: OIHW f32 -> [CoutPad][tap*Cin+ci] bf16 ----------
__global__ void pack_w(const float* __restrict__ src, u16* __restrict__ dst,
                       int Cin, int Cout, int CoutPad) {
  int total = CoutPad * 9 * Cin;
  for (int i = blockIdx.x * 256 + threadIdx.x; i < total; i += gridDim.x * 256) {
    int k = i % (9 * Cin);
    int n = i / (9 * Cin);
    int tap = k / Cin, ci = k % Cin;
    float v = (n < Cout) ? src[(n * Cin + ci) * 9 + tap] : 0.f;
    dst[i] = f32_to_bf16(v);
  }
}

// ---------- NCHW f32 -> NHWC bf16 (concat + nbr copy) ----------
__global__ void to_nhwc_cat(const float* __restrict__ nbr, const float* __restrict__ ref,
                            u16* __restrict__ xcat, u16* __restrict__ nbrh) {
  const long total = (long)BATCH * 128 * IMH * IMW;
  long i = (long)blockIdx.x * 256 + threadIdx.x;
  if (i >= total) return;
  int x = (int)(i % IMW);
  int y = (int)((i / IMW) % IMH);
  int c = (int)((i / ((long)IMW * IMH)) % 128);
  int b = (int)(i / ((long)IMW * IMH * 128));
  float v = (c < NF) ? nbr[(((long)b * NF + c) * IMH + y) * IMW + x]
                     : ref[(((long)b * NF + (c - NF)) * IMH + y) * IMW + x];
  u16 h = f32_to_bf16(v);
  xcat[((((long)b * IMH + y) * IMW + x) * 128) + c] = h;
  if (c < NF) nbrh[((((long)b * IMH + y) * IMW + x) * NF) + c] = h;
}

// ---------- implicit-GEMM 3x3 conv (stride1 pad1), NHWC bf16 -> NHWC ----------
// block: 256 thr (8 waves); tile 16(W) x 8(H); wave w = row w; K order = tap*CIN+c
template<int CIN, int COUT, int NT, bool RELU, bool F32OUT>
__global__ __launch_bounds__(256) void conv3x3_wmma(
    const u16* __restrict__ X, const u16* __restrict__ Wp,
    const float* __restrict__ bias, u16* __restrict__ Ybf, float* __restrict__ Yf) {
  constexpr int SITES = 10 * 18;
  __shared__ __align__(16) u16 tile[SITES * CIN];
  const int tid = threadIdx.x;
  const int x0 = blockIdx.x * 16, y0 = blockIdx.y * 8, b = blockIdx.z;

  // stage halo tile (zero-padded) into LDS; in-bounds units go direct-to-LDS async
  constexpr int C8 = CIN / 8;
  for (int i = tid; i < SITES * C8; i += 256) {
    int site = i / C8, c8 = i % C8;
    int r = site / 18, c = site % 18;
    int yy = y0 + r - 1, xx = x0 + c - 1;
    u16* dst = &tile[(long)site * CIN + c8 * 8];
    if (yy >= 0 && yy < IMH && xx >= 0 && xx < IMW) {
      async_copy_b128(&X[((((long)b * IMH + yy) * IMW + xx) * CIN) + c8 * 8], dst);
    } else {
      u32x4 z; z[0] = 0; z[1] = 0; z[2] = 0; z[3] = 0;
      *(u32x4*)dst = z;
    }
  }
  async_fence();
  __syncthreads();

  const int wv = tid >> 5, lane = tid & 31, lh = lane >> 4, ln = lane & 15;
  v8f acc[NT];
  #pragma unroll
  for (int i = 0; i < NT; ++i)
    #pragma unroll
    for (int j = 0; j < 8; ++j) acc[i][j] = 0.f;

  for (int tap = 0; tap < 9; ++tap) {
    const int dy = tap / 3, dx = tap % 3;
    const u16* arow = &tile[((wv + dy) * 18 + (ln + dx)) * CIN];
    #pragma unroll
    for (int cs = 0; cs < CIN / 32; ++cs) {
      Frag fa;
      const int c0 = cs * 32 + lh * 8;
      fa.q[0] = *(const u32x4*)(arow + c0);
      fa.q[1] = *(const u32x4*)(arow + c0 + 16);
      const int k0 = tap * CIN + cs * 32 + lh * 8;
      #pragma unroll
      for (int nt = 0; nt < NT; ++nt) {
        Frag fb;
        const u16* brow = &Wp[(long)(nt * 16 + ln) * (9 * CIN) + k0];
        fb.q[0] = *(const u32x4*)(brow);
        fb.q[1] = *(const u32x4*)(brow + 16);
        acc[nt] = __builtin_amdgcn_wmma_f32_16x16x32_bf16(
            false, fa.v, false, fb.v, (short)0, acc[nt], false, false);
      }
    }
  }

  // epilogue: bias (+ leaky relu), store NHWC
  const int y = y0 + wv;
  #pragma unroll
  for (int nt = 0; nt < NT; ++nt) {
    const int n = nt * 16 + ln;
    if (n >= COUT) continue;
    const float bv = bias[n];
    #pragma unroll
    for (int r = 0; r < 8; ++r) {
      const int x = x0 + r + lh * 8;
      float v = acc[nt][r] + bv;
      if (RELU) v = (v >= 0.f) ? v : 0.1f * v;
      const long o = (((long)b * IMH + y) * IMW + x);
      if constexpr (F32OUT) Yf[o * COUT + n] = v;
      else                  Ybf[o * COUT + n] = f32_to_bf16(v);
    }
  }
}

// ---------- deformable conv: sample->LDS per tap, then WMMA; out f32 NCHW ----------
__global__ __launch_bounds__(256) void dcn_wmma(
    const u16* __restrict__ Xn,     // nbr NHWC bf16 [B,H,W,64]
    const float* __restrict__ Off,  // NHWC f32 [B,H,W,72]
    const u16* __restrict__ Wp,     // [64][9*64] bf16
    const float* __restrict__ bias, float* __restrict__ Out) {
  __shared__ __align__(16) u16 S[8 * 16 * NF];   // 16 KB sampled A-tile
  const int tid = threadIdx.x;
  const int x0 = blockIdx.x * 16, y0 = blockIdx.y * 8, b = blockIdx.z;
  const int wv = tid >> 5, lane = tid & 31, lh = lane >> 4, ln = lane & 15;

  v8f acc[4];
  #pragma unroll
  for (int i = 0; i < 4; ++i)
    #pragma unroll
    for (int j = 0; j < 8; ++j) acc[i][j] = 0.f;

  for (int tap = 0; tap < 9; ++tap) {
    const int ky = tap / 3, kx = tap % 3;
    // bilinear sampling: 512 tasks = 128 px * 4 offset groups, 16 ch each
    for (int task = tid; task < 512; task += 256) {
      const int px = task & 127, g = task >> 7;
      const int yl = px >> 4, xl = px & 15;
      const int Y = y0 + yl, X = x0 + xl;
      const float* op = &Off[((((long)b * IMH + Y) * IMW + X) * OFFCH) + (g * 9 + tap) * 2];
      const float sy = (float)(Y - 1 + ky) + op[0];
      const float sx = (float)(X - 1 + kx) + op[1];
      const float fy = floorf(sy), fx = floorf(sx);
      const int iy = (int)fy, ix = (int)fx;
      const float ay = sy - fy, ax = sx - fx;
      float f[16];
      #pragma unroll
      for (int j = 0; j < 16; ++j) f[j] = 0.f;
      auto gather = [&](int cy, int cx, float wgt) {
        if (cy < 0 || cy >= IMH || cx < 0 || cx >= IMW) return;
        const u32* p = (const u32*)&Xn[((((long)b * IMH + cy) * IMW + cx) * NF) + g * 16];
        #pragma unroll
        for (int j = 0; j < 8; ++j) {
          const u32 u = p[j];
          f[2 * j]     += wgt * bf16lo(u);
          f[2 * j + 1] += wgt * bf16hi(u);
        }
      };
      gather(iy,     ix,     (1.f - ay) * (1.f - ax));
      gather(iy,     ix + 1, (1.f - ay) * ax);
      gather(iy + 1, ix,     ay * (1.f - ax));
      gather(iy + 1, ix + 1, ay * ax);
      u16* sp = &S[px * NF + g * 16];
      #pragma unroll
      for (int j = 0; j < 16; ++j) sp[j] = f32_to_bf16(f[j]);
    }
    __syncthreads();

    const u16* arow = &S[(wv * 16 + ln) * NF];
    #pragma unroll
    for (int cs = 0; cs < 2; ++cs) {
      Frag fa;
      const int c0 = cs * 32 + lh * 8;
      fa.q[0] = *(const u32x4*)(arow + c0);
      fa.q[1] = *(const u32x4*)(arow + c0 + 16);
      const int k0 = tap * NF + cs * 32 + lh * 8;
      #pragma unroll
      for (int nt = 0; nt < 4; ++nt) {
        Frag fb;
        const u16* brow = &Wp[(long)(nt * 16 + ln) * (9 * NF) + k0];
        fb.q[0] = *(const u32x4*)(brow);
        fb.q[1] = *(const u32x4*)(brow + 16);
        acc[nt] = __builtin_amdgcn_wmma_f32_16x16x32_bf16(
            false, fa.v, false, fb.v, (short)0, acc[nt], false, false);
      }
    }
    __syncthreads();
  }

  // epilogue: bias + leaky relu, f32 NCHW
  const int y = y0 + wv;
  #pragma unroll
  for (int nt = 0; nt < 4; ++nt) {
    const int n = nt * 16 + ln;
    const float bv = bias[n];
    #pragma unroll
    for (int r = 0; r < 8; ++r) {
      const int x = x0 + r + lh * 8;
      float v = acc[nt][r] + bv;
      v = (v >= 0.f) ? v : 0.1f * v;
      Out[(((long)b * NF + n) * IMH + y) * IMW + x] = v;
    }
  }
}

// ---------- host ----------
extern "C" void kernel_launch(void* const* d_in, const int* in_sizes, int n_in,
                              void* d_out, int out_size, void* d_ws, size_t ws_size,
                              hipStream_t stream) {
  const float* nbr   = (const float*)d_in[0];
  const float* ref   = (const float*)d_in[1];
  const float* w_co1 = (const float*)d_in[2];
  const float* b_co1 = (const float*)d_in[3];
  const float* w_m1  = (const float*)d_in[4];
  const float* b_m1  = (const float*)d_in[5];
  const float* w_m2  = (const float*)d_in[6];
  const float* b_m2  = (const float*)d_in[7];
  const float* w_m3  = (const float*)d_in[8];
  const float* b_m3  = (const float*)d_in[9];
  const float* w_dcn = (const float*)d_in[10];
  const float* b_dcn = (const float*)d_in[11];

  const size_t NP = (size_t)BATCH * IMH * IMW;   // 114688
  char* ws = (char*)d_ws;
  size_t cur = 0;
  auto alloc = [&](size_t bytes) -> void* {
    cur = (cur + 255) & ~(size_t)255;
    void* p = ws + cur;
    cur += bytes;
    return p;
  };
  u16* xcat  = (u16*)alloc(NP * 128 * 2);
  u16* nbrh  = (u16*)alloc(NP * NF * 2);
  u16* feat  = (u16*)alloc(NP * NF * 2);
  u16* t1    = (u16*)alloc(NP * NF * 2);
  u16* t2    = (u16*)alloc(NP * NF * 2);
  float* off = (float*)alloc(NP * OFFCH * 4);
  u16* wp1   = (u16*)alloc((size_t)64 * 9 * 128 * 2);
  u16* wpm1  = (u16*)alloc((size_t)64 * 9 * 64 * 2);
  u16* wpm2  = (u16*)alloc((size_t)64 * 9 * 64 * 2);
  u16* wpm3  = (u16*)alloc((size_t)80 * 9 * 64 * 2);   // padded to 80 N
  u16* wpd   = (u16*)alloc((size_t)64 * 9 * 64 * 2);

  // pack weights
  pack_w<<<288, 256, 0, stream>>>(w_co1, wp1, 128, 64, 64);
  pack_w<<<144, 256, 0, stream>>>(w_m1, wpm1, 64, 64, 64);
  pack_w<<<144, 256, 0, stream>>>(w_m2, wpm2, 64, 64, 64);
  pack_w<<<180, 256, 0, stream>>>(w_m3, wpm3, 64, OFFCH, 80);
  pack_w<<<144, 256, 0, stream>>>(w_dcn, wpd, 64, 64, 64);

  // activation layout conversion
  {
    const long total = (long)BATCH * 128 * IMH * IMW;
    to_nhwc_cat<<<(int)((total + 255) / 256), 256, 0, stream>>>(nbr, ref, xcat, nbrh);
  }

  dim3 gridc(IMW / 16, IMH / 8, BATCH);  // (14,16,4)
  // feat = lrelu(conv(cat, w_co1))
  conv3x3_wmma<128, 64, 4, true, false><<<gridc, 256, 0, stream>>>(xcat, wp1, b_co1, feat, nullptr);
  // offset head
  conv3x3_wmma<64, 64, 4, true, false><<<gridc, 256, 0, stream>>>(feat, wpm1, b_m1, t1, nullptr);
  conv3x3_wmma<64, 64, 4, true, false><<<gridc, 256, 0, stream>>>(t1, wpm2, b_m2, t2, nullptr);
  conv3x3_wmma<64, OFFCH, 5, false, true><<<gridc, 256, 0, stream>>>(t2, wpm3, b_m3, nullptr, off);
  // deformable conv + lrelu -> d_out (f32 NCHW)
  dcn_wmma<<<gridc, 256, 0, stream>>>(nbrh, off, wpd, b_dcn, (float*)d_out);
}
